// RAGFormer_17205638988679
// MI455X (gfx1250) — matrix operand order repaired and checked by reading, not compiled, not run on hardware
//
#include <hip/hip_runtime.h>
#include <hip/hip_bf16.h>
#include <math.h>

// ---------------------------------------------------------------------------
// Types for CDNA5 WMMA (wave32): bf16 A/B fragments (16 elems/lane = 8 VGPRs),
// f32 C/D accumulator (8 VGPRs).
// ---------------------------------------------------------------------------
typedef __bf16 bf16;
typedef __attribute__((ext_vector_type(16))) bf16  v16bf;
typedef __attribute__((ext_vector_type(8)))  bf16  v8bf;
typedef __attribute__((ext_vector_type(8)))  float v8f;

#define NN   4096
#define FF   512
#define DD   512
#define DFF_ 2048
#define LL   4
#define HH   8
#define DKK  64
#define ALPHA_C 0.1f
#define EPS_C   1e-5f

__device__ __forceinline__ bf16 to_bf16(float f) {
  union { float f; unsigned u; } a; a.f = f;
  unsigned r = a.u + 0x7FFFu + ((a.u >> 16) & 1u);   // round-nearest-even
  unsigned short hs = (unsigned short)(r >> 16);
  union { unsigned short s; bf16 b; } c; c.s = hs;
  return c.b;
}

// A fragment: 16x32 bf16 tile. lane 0-15: row M=lane, K = k0+kh*8..+7 (VGPR0-3)
// and K = k0+16+kh*8..+7 (VGPR4-7) with kh = lane>>4.   (ISA 7.12.2)
__device__ __forceinline__ v16bf load_afrag(const bf16* A, int lda, int m0,
                                            int k0, int lane) {
  const int m  = m0 + (lane & 15);
  const int kh = (lane >> 4) << 3;              // 0 or 8
  const bf16* p = A + (size_t)m * lda + k0 + kh;
  v8bf lo = *(const v8bf*)p;
  v8bf hi = *(const v8bf*)(p + 16);
  v16bf r;
#pragma unroll
  for (int i = 0; i < 8; ++i) { r[i] = lo[i]; r[i + 8] = hi[i]; }
  return r;
}

// B fragment: 32x16 bf16 tile of B = W^T, W row-major [Nout, K].
// lane & 15 = output column n; K = k0+kh*8..+7 and k0+16+kh*8..+7 of W row n
// with kh = lane>>4 (same K order as the A fragment) -> two 16-byte loads.
__device__ __forceinline__ v16bf load_bfrag(const bf16* W, int ldw, int n0,
                                            int k0, int lane) {
  const int n  = n0 + (lane & 15);
  const int kh = (lane >> 4) << 3;              // 0 or 8
  const bf16* p = W + (size_t)n * ldw + k0 + kh;
  v8bf lo = *(const v8bf*)p;
  v8bf hi = *(const v8bf*)(p + 16);
  v16bf r;
#pragma unroll
  for (int i = 0; i < 8; ++i) { r[i] = lo[i]; r[i + 8] = hi[i]; }
  return r;
}

__device__ __forceinline__ float gelu_exact(float x) {
  return 0.5f * x * (1.0f + erff(x * 0.70710678118654752f));
}

enum { EP_PLAIN = 0, EP_BIAS = 1, EP_GELU = 2, EP_RELU = 3, EP_RESID = 4, EP_PPR = 5 };

template <int MODE>
__device__ __forceinline__ void epilogue_elem(float v, int row, int col,
                                              const float* bias, const float* res,
                                              int ldr, float* Cf, int ldc,
                                              bf16* Cb, int ldcb, bf16* CbT,
                                              int ldcbt, float alpha) {
  if (MODE == EP_BIAS || MODE == EP_GELU || MODE == EP_RELU || MODE == EP_RESID) {
    if (bias) v += bias[col];
  }
  if (MODE == EP_GELU)  v = gelu_exact(v);
  if (MODE == EP_RELU)  v = fmaxf(v, 0.0f);
  if (MODE == EP_RESID) v += res[(size_t)row * ldr + col];
  if (MODE == EP_PPR)   v = (1.0f - alpha) * v + alpha * res[(size_t)row * ldr + col];
  if (Cf)  Cf[(size_t)row * ldc + col] = v;
  if (Cb)  Cb[(size_t)row * ldcb + col] = to_bf16(v);
  if (CbT) CbT[(size_t)col * ldcbt + row] = to_bf16(v);
}

// ---------------------------------------------------------------------------
// Main GEMM: C[M,Ntot] = A[M,K](bf16) @ W[Ntot,K]^T(bf16)  (+ epilogue)
// grid = (Ntot/128, M/128), block = 256 (8 waves laid out 4(M) x 2(N)).
// Wave tile 32(M)x64(N): 2 A fragments x 4 B fragments -> 8 WMMAs / K-step
// (1.5 b128 loads per WMMA; everything L2-resident on the 192MB L2).
// ---------------------------------------------------------------------------
template <int MODE>
__global__ __launch_bounds__(256)
void gemm_bt128(const bf16* A, int lda, const bf16* W, int ldw,
                const float* bias, const float* res, int ldr,
                float* Cf, int ldc, bf16* Cb, int ldcb, bf16* CbT, int ldcbt,
                int K, float alpha) {
  const int lane = threadIdx.x & 31;
  const int wave = threadIdx.x >> 5;
  const int n0 = blockIdx.x * 128 + (wave & 1) * 64;
  const int m0 = blockIdx.y * 128 + (wave >> 1) * 32;

  const v8f zero = {0.f, 0.f, 0.f, 0.f, 0.f, 0.f, 0.f, 0.f};
  v8f acc[2][4];
#pragma unroll
  for (int s = 0; s < 2; ++s)
#pragma unroll
    for (int t = 0; t < 4; ++t) acc[s][t] = zero;

  for (int k0 = 0; k0 < K; k0 += 32) {
    v16bf a0 = load_afrag(A, lda, m0, k0, lane);
    v16bf a1 = load_afrag(A, lda, m0 + 16, k0, lane);
#pragma unroll
    for (int t = 0; t < 4; ++t) {
      v16bf b = load_bfrag(W, ldw, n0 + 16 * t, k0, lane);
      acc[0][t] = __builtin_amdgcn_wmma_f32_16x16x32_bf16(
          false, a0, false, b, (short)0, acc[0][t], false, false);
      acc[1][t] = __builtin_amdgcn_wmma_f32_16x16x32_bf16(
          false, a1, false, b, (short)0, acc[1][t], false, false);
    }
  }

  const int colb = lane & 15;
  const int rofs = (lane >> 4) << 3;
#pragma unroll
  for (int s = 0; s < 2; ++s)
#pragma unroll
    for (int t = 0; t < 4; ++t)
#pragma unroll
      for (int r = 0; r < 8; ++r)
        epilogue_elem<MODE>(acc[s][t][r], m0 + s * 16 + rofs + r,
                            n0 + t * 16 + colb, bias, res, ldr, Cf, ldc, Cb,
                            ldcb, CbT, ldcbt, alpha);
}

// ---------------------------------------------------------------------------
// Narrow GEMM (Ntot == 64, used for attn @ V): wave tile 16x64,
// grid = (1, M/128), block = 256 (8 waves stacked in M).
// ---------------------------------------------------------------------------
template <int MODE>
__global__ __launch_bounds__(256)
void gemm_bt(const bf16* A, int lda, const bf16* W, int ldw,
             const float* bias, const float* res, int ldr,
             float* Cf, int ldc, bf16* Cb, int ldcb, bf16* CbT, int ldcbt,
             int K, float alpha) {
  const int lane = threadIdx.x & 31;
  const int wave = threadIdx.x >> 5;
  const int n0 = blockIdx.x * 64;
  const int m0 = blockIdx.y * 128 + wave * 16;

  const v8f zero = {0.f, 0.f, 0.f, 0.f, 0.f, 0.f, 0.f, 0.f};
  v8f acc[4];
#pragma unroll
  for (int t = 0; t < 4; ++t) acc[t] = zero;

  for (int k0 = 0; k0 < K; k0 += 32) {
    v16bf a = load_afrag(A, lda, m0, k0, lane);
#pragma unroll
    for (int t = 0; t < 4; ++t) {
      v16bf b = load_bfrag(W, ldw, n0 + 16 * t, k0, lane);
      acc[t] = __builtin_amdgcn_wmma_f32_16x16x32_bf16(
          false, a, false, b, (short)0, acc[t], false, false);
    }
  }

  const int colb = lane & 15;
  const int rofs = (lane >> 4) << 3;
#pragma unroll
  for (int t = 0; t < 4; ++t)
#pragma unroll
    for (int r = 0; r < 8; ++r)
      epilogue_elem<MODE>(acc[t][r], m0 + rofs + r, n0 + t * 16 + colb, bias,
                          res, ldr, Cf, ldc, Cb, ldcb, CbT, ldcbt, alpha);
}

// ---------------------------------------------------------------------------
// Support kernels
// ---------------------------------------------------------------------------
__global__ void f32_to_bf16_k(const float* s, bf16* d, size_t n) {
  size_t i = (size_t)blockIdx.x * blockDim.x + threadIdx.x;
  if (i < n) d[i] = to_bf16(s[i]);
}

// dst[c*R + r] = src[r*C + c]   (f32 -> bf16 transposed)
__global__ void transpose_to_bf16_k(const float* src, bf16* dst, int R, int C) {
  size_t idx = (size_t)blockIdx.x * blockDim.x + threadIdx.x;
  if (idx < (size_t)R * C) {
    int r = (int)(idx / C), c = (int)(idx % C);
    dst[(size_t)c * R + r] = to_bf16(src[idx]);
  }
}

__global__ void cat_k(const float* x0, const float* h, float* catf, bf16* catb) {
  size_t idx = (size_t)blockIdx.x * blockDim.x + threadIdx.x;
  const size_t total = (size_t)NN * 2 * FF;
  if (idx < total) {
    int row = (int)(idx / (2 * FF));
    int col = (int)(idx % (2 * FF));
    float v = (col < FF) ? x0[(size_t)row * FF + col]
                         : h[(size_t)row * FF + (col - FF)];
    catf[idx] = v;
    catb[idx] = to_bf16(v);
  }
}

__global__ __launch_bounds__(256)
void layernorm_k(const float* x, const float* g, const float* b,
                 bf16* yb, float* yf, int Dm) {
  const int row = blockIdx.x;
  const int tid = threadIdx.x;
  const float* xr = x + (size_t)row * Dm;
  float s = 0.f, s2 = 0.f;
  for (int i = tid; i < Dm; i += 256) { float v = xr[i]; s += v; s2 += v * v; }
  __shared__ float sh0[256], sh1[256];
  sh0[tid] = s; sh1[tid] = s2;
  __syncthreads();
  for (int off = 128; off > 0; off >>= 1) {
    if (tid < off) { sh0[tid] += sh0[tid + off]; sh1[tid] += sh1[tid + off]; }
    __syncthreads();
  }
  const float mean = sh0[0] / Dm;
  const float var  = sh1[0] / Dm - mean * mean;
  const float inv  = rsqrtf(var + EPS_C);
  for (int i = tid; i < Dm; i += 256) {
    float v = (xr[i] - mean) * inv * g[i] + b[i];
    yb[(size_t)row * Dm + i] = to_bf16(v);
    if (yf) yf[(size_t)row * Dm + i] = v;
  }
}

__global__ __launch_bounds__(256)
void softmax_bf16_k(const float* S, bf16* P, int Nc) {
  const int row = blockIdx.x;
  const int tid = threadIdx.x;
  const float* sr = S + (size_t)row * Nc;
  __shared__ float sh[256];
  float mx = -3.4e38f;
  for (int i = tid; i < Nc; i += 256) mx = fmaxf(mx, sr[i]);
  sh[tid] = mx; __syncthreads();
  for (int off = 128; off > 0; off >>= 1) {
    if (tid < off) sh[tid] = fmaxf(sh[tid], sh[tid + off]);
    __syncthreads();
  }
  mx = sh[0]; __syncthreads();
  float sum = 0.f;
  for (int i = tid; i < Nc; i += 256) sum += __expf(sr[i] - mx);
  sh[tid] = sum; __syncthreads();
  for (int off = 128; off > 0; off >>= 1) {
    if (tid < off) sh[tid] += sh[tid + off];
    __syncthreads();
  }
  const float inv = 1.0f / sh[0];
  for (int i = tid; i < Nc; i += 256)
    P[(size_t)row * Nc + i] = to_bf16(__expf(sr[i] - mx) * inv);
}

__global__ __launch_bounds__(256)
void err_k(const float* recon, const float* catf, float* err, int Cm) {
  const int row = blockIdx.x;
  const int tid = threadIdx.x;
  __shared__ float sh[256];
  float s = 0.f;
  for (int i = tid; i < Cm; i += 256) {
    float d = recon[(size_t)row * Cm + i] - catf[(size_t)row * Cm + i];
    s += d * d;
  }
  sh[tid] = s; __syncthreads();
  for (int off = 128; off > 0; off >>= 1) {
    if (tid < off) sh[tid] += sh[tid + off];
    __syncthreads();
  }
  if (tid == 0) err[row] = sh[0] / Cm;
}

__global__ __launch_bounds__(1024)
void minmax_k(const float* err, float* red, int n) {
  const int tid = threadIdx.x;
  __shared__ float smn[1024], smx[1024];
  float mn = 3.4e38f, mx = -3.4e38f;
  for (int i = tid; i < n; i += 1024) {
    mn = fminf(mn, err[i]); mx = fmaxf(mx, err[i]);
  }
  smn[tid] = mn; smx[tid] = mx; __syncthreads();
  for (int off = 512; off > 0; off >>= 1) {
    if (tid < off) {
      smn[tid] = fminf(smn[tid], smn[tid + off]);
      smx[tid] = fmaxf(smx[tid], smx[tid + off]);
    }
    __syncthreads();
  }
  if (tid == 0) { red[0] = smn[0]; red[1] = smx[0]; }
}

__global__ void norm_k(const float* err, const float* red, float* out, int n) {
  int i = blockIdx.x * blockDim.x + threadIdx.x;
  if (i < n) {
    float mn = red[0], mx = red[1];
    out[i] = (mx > mn) ? (err[i] - mn) / (mx - mn) : 0.5f;
  }
}

// ---------------------------------------------------------------------------
// Host orchestration
// ---------------------------------------------------------------------------
extern "C" void kernel_launch(void* const* d_in, const int* in_sizes, int n_in,
                              void* d_out, int out_size, void* d_ws, size_t ws_size,
                              hipStream_t stream) {
  const float* x0   = (const float*)d_in[0];
  const float* adj  = (const float*)d_in[1];
  const float* Wp   = (const float*)d_in[2];
  const float* bp   = (const float*)d_in[3];
  const float* ln1g = (const float*)d_in[4];
  const float* ln1b = (const float*)d_in[5];
  const float* Wq   = (const float*)d_in[6];
  const float* bq   = (const float*)d_in[7];
  const float* Wk   = (const float*)d_in[8];
  const float* bk   = (const float*)d_in[9];
  const float* Wv   = (const float*)d_in[10];
  const float* bv   = (const float*)d_in[11];
  const float* Wo   = (const float*)d_in[12];
  const float* bo   = (const float*)d_in[13];
  const float* ln2g = (const float*)d_in[14];
  const float* ln2b = (const float*)d_in[15];
  const float* W1   = (const float*)d_in[16];
  const float* b1   = (const float*)d_in[17];
  const float* W2   = (const float*)d_in[18];
  const float* b2   = (const float*)d_in[19];
  const float* lnfg = (const float*)d_in[20];
  const float* lnfb = (const float*)d_in[21];
  const float* Wd1  = (const float*)d_in[22];
  const float* bd1  = (const float*)d_in[23];
  const float* Wd2  = (const float*)d_in[24];
  const float* bd2  = (const float*)d_in[25];
  float* outF = (float*)d_out;

  // ---- workspace layout (256B aligned) ----
  char* base = (char*)d_ws;
  size_t off = 0;
  auto alc = [&](size_t bytes) -> char* {
    char* p = base + off;
    off = (off + bytes + 255) & ~(size_t)255;
    return p;
  };
  bf16*  adjb  = (bf16*) alc((size_t)NN * NN * 2);
  float* h     = (float*)alc((size_t)NN * FF * 4);
  bf16*  hT    = (bf16*) alc((size_t)FF * NN * 2);
  float* catf  = (float*)alc((size_t)NN * 2 * FF * 4);
  bf16*  catb  = (bf16*) alc((size_t)NN * 2 * FF * 2);
  float* emb   = (float*)alc((size_t)NN * DD * 4);
  bf16*  yb    = (bf16*) alc((size_t)NN * DD * 2);
  bf16*  qb    = (bf16*) alc((size_t)NN * DD * 2);
  bf16*  kb    = (bf16*) alc((size_t)NN * DD * 2);
  bf16*  vTb   = (bf16*) alc((size_t)DD * NN * 2);
  bf16*  ob    = (bf16*) alc((size_t)NN * DD * 2);
  bf16*  ffb   = (bf16*) alc((size_t)NN * DFF_ * 2);
  float* score = (float*)alc((size_t)NN * NN * 4);
  bf16*  attnb = (bf16*) alc((size_t)NN * NN * 2);
  float* recon = score;  // reuse 64MB region after attention is done
  float* errv  = (float*)alc((size_t)NN * 4);
  float* red   = (float*)alc(256);
  bf16*  Wpb   = (bf16*) alc((size_t)DD * 2 * FF * 2);
  bf16*  Wqb   = (bf16*) alc((size_t)LL * DD * DD * 2);
  bf16*  Wkb   = (bf16*) alc((size_t)LL * DD * DD * 2);
  bf16*  Wvb   = (bf16*) alc((size_t)LL * DD * DD * 2);
  bf16*  Wob   = (bf16*) alc((size_t)LL * DD * DD * 2);
  bf16*  W1b   = (bf16*) alc((size_t)LL * DFF_ * DD * 2);
  bf16*  W2b   = (bf16*) alc((size_t)LL * DD * DFF_ * 2);
  bf16*  Wd1b  = (bf16*) alc((size_t)DD * DD * 2);
  bf16*  Wd2b  = (bf16*) alc((size_t)2 * FF * DD * 2);

  auto conv = [&](const float* s, bf16* d, size_t n) {
    f32_to_bf16_k<<<dim3((unsigned)((n + 255) / 256)), 256, 0, stream>>>(s, d, n);
  };

  // ---- bf16 operand conversions ----
  conv(adj, adjb, (size_t)NN * NN);
  conv(Wp,  Wpb,  (size_t)DD * 2 * FF);
  conv(Wq,  Wqb,  (size_t)LL * DD * DD);
  conv(Wk,  Wkb,  (size_t)LL * DD * DD);
  conv(Wv,  Wvb,  (size_t)LL * DD * DD);
  conv(Wo,  Wob,  (size_t)LL * DD * DD);
  conv(W1,  W1b,  (size_t)LL * DFF_ * DD);
  conv(W2,  W2b,  (size_t)LL * DD * DFF_);
  conv(Wd1, Wd1b, (size_t)DD * DD);
  conv(Wd2, Wd2b, (size_t)2 * FF * DD);

  // ---- PPR propagation: h = x0; 3x { h = 0.9*(adj @ h) + 0.1*x0 } ----
  hipMemcpyAsync(h, x0, (size_t)NN * FF * 4, hipMemcpyDeviceToDevice, stream);
  for (int s = 0; s < 3; ++s) {
    transpose_to_bf16_k<<<dim3((NN * FF + 255) / 256), 256, 0, stream>>>(h, hT, NN, FF);
    gemm_bt128<EP_PPR><<<dim3(FF / 128, NN / 128), 256, 0, stream>>>(
        adjb, NN, hT, NN, nullptr, x0, FF, h, FF, nullptr, 0, nullptr, 0, NN, ALPHA_C);
  }

  // ---- concat + token projection: emb = cat @ Wp^T + bp ----
  cat_k<<<dim3((NN * 2 * FF + 255) / 256), 256, 0, stream>>>(x0, h, catf, catb);
  gemm_bt128<EP_BIAS><<<dim3(DD / 128, NN / 128), 256, 0, stream>>>(
      catb, 2 * FF, Wpb, 2 * FF, bp, nullptr, 0, emb, DD, nullptr, 0, nullptr, 0,
      2 * FF, 0.f);

  // ---- transformer layers ----
  for (int l = 0; l < LL; ++l) {
    layernorm_k<<<NN, 256, 0, stream>>>(emb, ln1g + l * DD, ln1b + l * DD, yb, nullptr, DD);

    gemm_bt128<EP_BIAS><<<dim3(DD / 128, NN / 128), 256, 0, stream>>>(
        yb, DD, Wqb + (size_t)l * DD * DD, DD, bq + l * DD, nullptr, 0,
        nullptr, 0, qb, DD, nullptr, 0, DD, 0.f);
    gemm_bt128<EP_BIAS><<<dim3(DD / 128, NN / 128), 256, 0, stream>>>(
        yb, DD, Wkb + (size_t)l * DD * DD, DD, bk + l * DD, nullptr, 0,
        nullptr, 0, kb, DD, nullptr, 0, DD, 0.f);
    // V: store transposed bf16 [DD, NN] so attn@v is in W^T form
    gemm_bt128<EP_BIAS><<<dim3(DD / 128, NN / 128), 256, 0, stream>>>(
        yb, DD, Wvb + (size_t)l * DD * DD, DD, bv + l * DD, nullptr, 0,
        nullptr, 0, nullptr, 0, vTb, NN, DD, 0.f);

    for (int hd = 0; hd < HH; ++hd) {
      // scores = Q_h @ K_h^T    (scale = 1 per reference)
      gemm_bt128<EP_PLAIN><<<dim3(NN / 128, NN / 128), 256, 0, stream>>>(
          qb + hd * DKK, DD, kb + hd * DKK, DD, nullptr, nullptr, 0,
          score, NN, nullptr, 0, nullptr, 0, DKK, 0.f);
      softmax_bf16_k<<<NN, 256, 0, stream>>>(score, attnb, NN);
      // O_h = attn @ V_h   (via V^T rows); Ntot = 64 -> narrow kernel
      gemm_bt<EP_PLAIN><<<dim3(1, NN / 128), 256, 0, stream>>>(
          attnb, NN, vTb + (size_t)hd * DKK * NN, NN, nullptr, nullptr, 0,
          nullptr, 0, ob + hd * DKK, DD, nullptr, 0, NN, 0.f);
    }

    // emb += O @ Wo^T + bo
    gemm_bt128<EP_RESID><<<dim3(DD / 128, NN / 128), 256, 0, stream>>>(
        ob, DD, Wob + (size_t)l * DD * DD, DD, bo + l * DD, emb, DD,
        emb, DD, nullptr, 0, nullptr, 0, DD, 0.f);

    // FFN
    layernorm_k<<<NN, 256, 0, stream>>>(emb, ln2g + l * DD, ln2b + l * DD, yb, nullptr, DD);
    gemm_bt128<EP_GELU><<<dim3(DFF_ / 128, NN / 128), 256, 0, stream>>>(
        yb, DD, W1b + (size_t)l * DFF_ * DD, DD, b1 + l * DFF_, nullptr, 0,
        nullptr, 0, ffb, DFF_, nullptr, 0, DD, 0.f);
    gemm_bt128<EP_RESID><<<dim3(DD / 128, NN / 128), 256, 0, stream>>>(
        ffb, DFF_, W2b + (size_t)l * DD * DFF_, DFF_, b2 + l * DD, emb, DD,
        emb, DD, nullptr, 0, nullptr, 0, DFF_, 0.f);
  }

  // ---- final LN (f32 result is output[0]) ----
  layernorm_k<<<NN, 256, 0, stream>>>(emb, lnfg, lnfb, yb, outF, DD);

  // ---- decoder + anomaly scores ----
  gemm_bt128<EP_RELU><<<dim3(DD / 128, NN / 128), 256, 0, stream>>>(
      yb, DD, Wd1b, DD, bd1, nullptr, 0, nullptr, 0, qb /*reuse*/, DD,
      nullptr, 0, DD, 0.f);
  gemm_bt128<EP_BIAS><<<dim3(2 * FF / 128, NN / 128), 256, 0, stream>>>(
      qb, DD, Wd2b, DD, bd2, nullptr, 0, recon, 2 * FF, nullptr, 0, nullptr, 0,
      DD, 0.f);

  err_k<<<NN, 256, 0, stream>>>(recon, catf, errv, 2 * FF);
  minmax_k<<<1, 1024, 0, stream>>>(errv, red, NN);
  norm_k<<<dim3((NN + 255) / 256), 256, 0, stream>>>(errv, red, outF + (size_t)NN * DD, NN);
}
